// SimpleTransformerBlock_49031346651644
// MI455X (gfx1250) — compile-verified, matrix operand order checked by reading, not run on hardware
//
#include <hip/hip_runtime.h>
#include <hip/hip_bf16.h>
#include <math.h>

typedef _Float16 f16;
typedef __attribute__((ext_vector_type(16))) _Float16 v16h;
typedef __attribute__((ext_vector_type(8)))  _Float16 v8h;
typedef __attribute__((ext_vector_type(8)))  float    v8f;
typedef __attribute__((ext_vector_type(2)))  _Float16 h2;

#define D_MODEL 1024
#define D_FF    4096
#define NHEAD   16
#define HDIM    64
#define BATCH   2
#define SEQ     2048
#define MROWS   (BATCH*SEQ)   // 4096

// ---------------------------------------------------------------------------
// CDNA5 WMMA 16x16x32 f16 layouts (ISA 7.12.2, wave32):
// A (16x32): lane&15 = row M; per lane the 16 fragment f16s are two contiguous
// 16B chunks of the 64B K-row at bytes hf*16 and 32+hf*16  -> 2 x b128 loads.
// B (32x16): lane&15 = col N; same K split (read from a [n][k] panel).
// C/D f32:   element v <-> (row = v + 8*hf, col = lane&15)
// ---------------------------------------------------------------------------
__device__ __forceinline__ v16h frag_ld(const f16* p) {
  int hf = (threadIdx.x >> 4) & 1;
  v8h lo = *(const v8h*)(p + hf * 8);
  v8h hi = *(const v8h*)(p + 16 + hf * 8);
  return __builtin_shufflevector(lo, hi, 0, 1, 2, 3, 4, 5, 6, 7,
                                 8, 9, 10, 11, 12, 13, 14, 15);
}

__device__ __forceinline__ v16h a_frag_load(const f16* A, long lda, int row0, int k0) {
  int lr = threadIdx.x & 15;
  return frag_ld(A + (long)(row0 + lr) * lda + k0);
}

// B-frag from a [col][k] panel in LDS, row stride ldp (multiple of 8 f16)
__device__ __forceinline__ v16h b_frag_lds(const f16* Pt, int ldp, int col0, int kc) {
  int lr = threadIdx.x & 15;
  return frag_ld(Pt + (long)(col0 + lr) * ldp + kc);
}

// ---------------------------------------------------------------------------
// CDNA5 async global -> LDS (ASYNCcnt-tracked DMA; ISA 10.x / 15.18.3)
// ---------------------------------------------------------------------------
__device__ __forceinline__ void async_ld_b128(const void* g, void* lds) {
  unsigned long long ga = (unsigned long long)g;
  unsigned int la = (unsigned int)(unsigned long long)lds;
  asm volatile("global_load_async_to_lds_b128 %0, %1, off"
               :: "v"(la), "v"(ga) : "memory");
}
__device__ __forceinline__ void async_wait0() {
  asm volatile("s_wait_asynccnt 0" ::: "memory");
}

// ---------------------------------------------------------------------------
// fp32 -> fp16 convert + transpose (weights -> Wt[n][k])
// ---------------------------------------------------------------------------
__global__ void cvt_tr_f16_kernel(const float* __restrict__ in, f16* __restrict__ out,
                                  int K, int N) {   // in[K][N] -> out[N][K]
  long i = (long)blockIdx.x * blockDim.x + threadIdx.x;
  if (i < (long)K * N) {
    int n = (int)(i / K), k = (int)(i % K);
    out[i] = (f16)in[(long)k * N + n];
  }
}

// ---------------------------------------------------------------------------
// LayerNorm (f32 in) -> f16 out, one row (D_MODEL) per block
// ---------------------------------------------------------------------------
__global__ __launch_bounds__(256) void ln_f16_kernel(
    const float* __restrict__ X, const float* __restrict__ g,
    const float* __restrict__ be, f16* __restrict__ out, int D) {
  __shared__ float red[256];
  int row = blockIdx.x;
  const float* x = X + (long)row * D;
  float s = 0.f;
  for (int i = threadIdx.x; i < D; i += 256) s += x[i];
  red[threadIdx.x] = s; __syncthreads();
  for (int st = 128; st > 0; st >>= 1) {
    if (threadIdx.x < st) red[threadIdx.x] += red[threadIdx.x + st];
    __syncthreads();
  }
  float mu = red[0] / D; __syncthreads();
  float vs = 0.f;
  for (int i = threadIdx.x; i < D; i += 256) { float d = x[i] - mu; vs += d * d; }
  red[threadIdx.x] = vs; __syncthreads();
  for (int st = 128; st > 0; st >>= 1) {
    if (threadIdx.x < st) red[threadIdx.x] += red[threadIdx.x + st];
    __syncthreads();
  }
  float rstd = rsqrtf(red[0] / D + 1e-5f);
  for (int i = threadIdx.x; i < D; i += 256)
    out[(long)row * D + i] = (f16)((x[i] - mu) * rstd * g[i] + be[i]);
}

// ---------------------------------------------------------------------------
// WMMA GEMM: C[M,N] = A[M,K](f16) * W(f16, pre-transposed Wt[N][K]) + bias.
// Block = 8 waves, each a 32x64 tile -> block 256x64. The 64x64 B panel is
// DMA'd into LDS with async-to-LDS loads and DOUBLE BUFFERED: the next panel
// streams in while the current one feeds 16 WMMAs.
// Epilogue: 0 f16 row-major | 1 f16 Q/K [B,H,S,Dh] | 2 f32 res+acc+bias
//           3 f16 GELU      | 4 f16 V [B,H,Dh,S]
// ---------------------------------------------------------------------------
#define BT_LD 72   // 64 + 8 pad: rows stay 16B aligned

__global__ __launch_bounds__(256) void gemm_wmma_kernel(
    const f16* __restrict__ A, const f16* __restrict__ Wt,
    const float* __restrict__ bias, const float* __restrict__ res,
    void* __restrict__ out, int M, int N, int K, int mode, int Sremap) {
  __shared__ f16 Bt[2][64][BT_LD];     // double-buffered [n][k] panel
  int tid = threadIdx.x;
  int w = tid >> 5;
  int mtile = blockIdx.y * 256 + w * 32;
  int n0 = blockIdx.x * 64;
  int l = tid & 31, lr = l & 15, hf = l >> 4;

  // staging coords: thread -> (n row, 16-k chunk) of the 64x64 panel
  int sn  = tid >> 2;           // 0..63
  int skc = (tid & 3) * 16;     // 0,16,32,48

  // prologue: DMA panel 0 into buffer 0
  {
    const f16* src = Wt + (long)(n0 + sn) * K + skc;
    async_ld_b128(src,     &Bt[0][sn][skc]);
    async_ld_b128(src + 8, &Bt[0][sn][skc + 8]);
  }
  async_wait0();
  __syncthreads();

  v8f acc[2][4] = {};
  for (int k0 = 0; k0 < K; k0 += 64) {
    int buf = (k0 >> 6) & 1;
    // kick off DMA of the next panel into the other buffer
    if (k0 + 64 < K) {
      const f16* src = Wt + (long)(n0 + sn) * K + k0 + 64 + skc;
      async_ld_b128(src,     &Bt[buf ^ 1][sn][skc]);
      async_ld_b128(src + 8, &Bt[buf ^ 1][sn][skc + 8]);
    }

#pragma unroll
    for (int kc = 0; kc < 64; kc += 32) {
      v16h af0 = a_frag_load(A, K, mtile, k0 + kc);
      v16h af1 = a_frag_load(A, K, mtile + 16, k0 + kc);
#pragma unroll
      for (int nt = 0; nt < 4; ++nt) {
        v16h bf = b_frag_lds(&Bt[buf][0][0], BT_LD, nt * 16, kc);
        acc[0][nt] = __builtin_amdgcn_wmma_f32_16x16x32_f16(
            false, af0, false, bf, (short)0, acc[0][nt], false, false);
        acc[1][nt] = __builtin_amdgcn_wmma_f32_16x16x32_f16(
            false, af1, false, bf, (short)0, acc[1][nt], false, false);
      }
    }
    async_wait0();
    __syncthreads();
  }

#pragma unroll
  for (int r = 0; r < 2; ++r) {
#pragma unroll
    for (int nt = 0; nt < 4; ++nt) {
      int col = n0 + nt * 16 + lr;
      float bv = bias ? bias[col] : 0.f;
#pragma unroll
      for (int v = 0; v < 8; ++v) {
        int row = mtile + r * 16 + v + 8 * hf;
        float val = acc[r][nt][v] + bv;
        if (mode == 0) {
          ((f16*)out)[(long)row * N + col] = (f16)val;
        } else if (mode == 1) {
          int b = row / Sremap, s = row % Sremap;
          int head = col >> 6, d = col & 63;
          ((f16*)out)[(((long)(b * NHEAD + head) * Sremap + s) << 6) + d] = (f16)val;
        } else if (mode == 2) {
          long idx = (long)row * N + col;
          ((float*)out)[idx] = res[idx] + val;
        } else if (mode == 3) {
          float gl = 0.5f * val * (1.0f + erff(val * 0.70710678118f));
          ((f16*)out)[(long)row * N + col] = (f16)gl;
        } else {  // mode 4: V in [B,H,Dh,S]
          int b = row / Sremap, s = row % Sremap;
          int head = col >> 6, d = col & 63;
          ((f16*)out)[((long)(b * NHEAD + head) * HDIM + d) * Sremap + s] = (f16)val;
        }
      }
    }
  }
}

// ---------------------------------------------------------------------------
// Flash attention. 8 waves share one (b,h); per 32-key step the K tile
// (Kt[key][dim], from [B,H,S,Dh]) and V tile (Vt[dim][key], from the
// pre-transposed [B,H,Dh,S] buffer) are DMA'd into LDS with async-to-LDS.
// P (C-layout) -> A-layout via per-wave LDS bounce; online softmax via
// wave32 half-shuffles. O: [B,S,D_MODEL] f16.
// ---------------------------------------------------------------------------
#define KT_LD 72
#define VT_LD 40

__global__ __launch_bounds__(256) void flash_attn_kernel(
    const f16* __restrict__ Q, const f16* __restrict__ Kb,
    const f16* __restrict__ Vt_g, f16* __restrict__ O, int S) {
  __shared__ f16 Kt[32][KT_LD];
  __shared__ f16 Vt[64][VT_LD];
  __shared__ f16 pS[8][16][32];
  int tid = threadIdx.x;
  int w = tid >> 5;
  int l = tid & 31, lr = l & 15, hf = l >> 4;
  int bh = blockIdx.x;                 // b*NHEAD + head
  int b = bh >> 4, head = bh & 15;
  int qtile = (blockIdx.y * 8 + w) * 16;
  const f16* Qp = Q + (long)bh * S * HDIM;
  const f16* Kp = Kb + (long)bh * S * HDIM;
  const f16* Vp = Vt_g + (long)bh * HDIM * S;   // [Dh][S]

  v16h qf0 = a_frag_load(Qp, HDIM, qtile, 0);
  v16h qf1 = a_frag_load(Qp, HDIM, qtile, 32);

  // staging coords
  int sk  = tid >> 3;           // 0..31 key (Kt)
  int sc  = (tid & 7) * 8;      // 0..56 dim chunk (Kt)
  int sd  = tid >> 2;           // 0..63 dim (Vt)
  int skc = (tid & 3) * 8;      // 0,8,16,24 key chunk (Vt)

  float mrow[8], lrow[8];
  v8f oacc[4] = {};
#pragma unroll
  for (int v = 0; v < 8; ++v) { mrow[v] = -1e30f; lrow[v] = 0.f; }

  for (int kt = 0; kt < S; kt += 32) {
    // ---- DMA K and V tiles into LDS ----
    async_ld_b128(Kp + (long)(kt + sk) * HDIM + sc, &Kt[sk][sc]);
    async_ld_b128(Vp + (long)sd * S + kt + skc,     &Vt[sd][skc]);
    async_wait0();
    __syncthreads();

    // ---- scores: S = Q.K^T * 1/sqrt(Dh), two 16-key subtiles ----
    float sv[2][8];
    {
      v16h ka0 = b_frag_lds(&Kt[0][0],  KT_LD, 0, 0);
      v16h ka1 = b_frag_lds(&Kt[0][0],  KT_LD, 0, 32);
      v16h kb0 = b_frag_lds(&Kt[16][0], KT_LD, 0, 0);
      v16h kb1 = b_frag_lds(&Kt[16][0], KT_LD, 0, 32);
      v8f c0 = {}, c1 = {};
      c0 = __builtin_amdgcn_wmma_f32_16x16x32_f16(false, qf0, false, ka0, (short)0, c0, false, false);
      c0 = __builtin_amdgcn_wmma_f32_16x16x32_f16(false, qf1, false, ka1, (short)0, c0, false, false);
      c1 = __builtin_amdgcn_wmma_f32_16x16x32_f16(false, qf0, false, kb0, (short)0, c1, false, false);
      c1 = __builtin_amdgcn_wmma_f32_16x16x32_f16(false, qf1, false, kb1, (short)0, c1, false, false);
#pragma unroll
      for (int v = 0; v < 8; ++v) { sv[0][v] = c0[v] * 0.125f; sv[1][v] = c1[v] * 0.125f; }
    }

    // ---- online softmax (rows live across a 16-lane half) ----
    float alpha[8];
#pragma unroll
    for (int v = 0; v < 8; ++v) {
      float tmax = fmaxf(sv[0][v], sv[1][v]);
      for (int off = 8; off >= 1; off >>= 1)
        tmax = fmaxf(tmax, __shfl_xor(tmax, off, 32));
      float mnew = fmaxf(mrow[v], tmax);
      alpha[v] = __expf(mrow[v] - mnew);
      mrow[v] = mnew;
      float p0 = __expf(sv[0][v] - mnew);
      float p1 = __expf(sv[1][v] - mnew);
      float rs = p0 + p1;
      for (int off = 8; off >= 1; off >>= 1)
        rs += __shfl_xor(rs, off, 32);
      lrow[v] = lrow[v] * alpha[v] + rs;
      pS[w][v + 8 * hf][lr]      = (f16)p0;   // C-layout -> LDS
      pS[w][v + 8 * hf][16 + lr] = (f16)p1;
    }
    asm volatile("s_wait_dscnt 0" ::: "memory");  // same-wave LDS RAW

    // ---- P as A-frag, V as 4 B-frags, then 4 WMMAs ----
    v16h pf  = frag_ld(&pS[w][lr][0]);
    v16h vb0 = b_frag_lds(&Vt[0][0],  VT_LD, 0, 0);
    v16h vb1 = b_frag_lds(&Vt[16][0], VT_LD, 0, 0);
    v16h vb2 = b_frag_lds(&Vt[32][0], VT_LD, 0, 0);
    v16h vb3 = b_frag_lds(&Vt[48][0], VT_LD, 0, 0);
#pragma unroll
    for (int nt = 0; nt < 4; ++nt)
#pragma unroll
      for (int v = 0; v < 8; ++v) oacc[nt][v] *= alpha[v];
    oacc[0] = __builtin_amdgcn_wmma_f32_16x16x32_f16(false, pf, false, vb0, (short)0, oacc[0], false, false);
    oacc[1] = __builtin_amdgcn_wmma_f32_16x16x32_f16(false, pf, false, vb1, (short)0, oacc[1], false, false);
    oacc[2] = __builtin_amdgcn_wmma_f32_16x16x32_f16(false, pf, false, vb2, (short)0, oacc[2], false, false);
    oacc[3] = __builtin_amdgcn_wmma_f32_16x16x32_f16(false, pf, false, vb3, (short)0, oacc[3], false, false);
    __syncthreads();   // protect Kt/Vt before next staging
  }

#pragma unroll
  for (int nt = 0; nt < 4; ++nt) {
    int col = head * HDIM + nt * 16 + lr;
#pragma unroll
    for (int v = 0; v < 8; ++v) {
      int q = qtile + v + 8 * hf;
      O[((long)(b * S + q)) * D_MODEL + col] = (f16)(oacc[nt][v] / lrow[v]);
    }
  }
}

// ---------------------------------------------------------------------------
// Host orchestration
// ---------------------------------------------------------------------------
extern "C" void kernel_launch(void* const* d_in, const int* in_sizes, int n_in,
                              void* d_out, int out_size, void* d_ws, size_t ws_size,
                              hipStream_t stream) {
  const float* x  = (const float*)d_in[0];
  const float* wq = (const float*)d_in[1];  const float* bq = (const float*)d_in[2];
  const float* wk = (const float*)d_in[3];  const float* bk = (const float*)d_in[4];
  const float* wv = (const float*)d_in[5];  const float* bv = (const float*)d_in[6];
  const float* wo = (const float*)d_in[7];  const float* bo = (const float*)d_in[8];
  const float* w1 = (const float*)d_in[9];  const float* b1 = (const float*)d_in[10];
  const float* w2 = (const float*)d_in[11]; const float* b2 = (const float*)d_in[12];
  const float* g1 = (const float*)d_in[13]; const float* be1 = (const float*)d_in[14];
  const float* g2 = (const float*)d_in[15]; const float* be2 = (const float*)d_in[16];
  float* out = (float*)d_out;

  char* ws = (char*)d_ws;
  const long DD  = (long)D_MODEL * D_MODEL;
  const long DF  = (long)D_MODEL * D_FF;
  f16*  wq_t = (f16*)(ws);                      // all weights transposed [N][K]
  f16*  wk_t = (f16*)(ws + 2u  * 1024 * 1024);
  f16*  wv_t = (f16*)(ws + 4u  * 1024 * 1024);
  f16*  wo_t = (f16*)(ws + 6u  * 1024 * 1024);
  f16*  w1_t = (f16*)(ws + 8u  * 1024 * 1024);
  f16*  w2_t = (f16*)(ws + 16u * 1024 * 1024);
  f16*  h1   = (f16*)(ws + 24u * 1024 * 1024);
  f16*  qb   = (f16*)(ws + 32u * 1024 * 1024);
  f16*  kb   = (f16*)(ws + 40u * 1024 * 1024);
  f16*  vt   = (f16*)(ws + 48u * 1024 * 1024);  // V as [B,H,Dh,S]
  f16*  ob   = (f16*)(ws + 56u * 1024 * 1024);
  float* x2  = (float*)(ws + 64u * 1024 * 1024);
  f16*  h2b  = (f16*)(ws + 80u * 1024 * 1024);
  f16*  ffn  = (f16*)(ws + 88u * 1024 * 1024);

  // weight conversion + transpose (f32 [K][N] -> f16 [N][K])
  cvt_tr_f16_kernel<<<(DD + 255) / 256, 256, 0, stream>>>(wq, wq_t, D_MODEL, D_MODEL);
  cvt_tr_f16_kernel<<<(DD + 255) / 256, 256, 0, stream>>>(wk, wk_t, D_MODEL, D_MODEL);
  cvt_tr_f16_kernel<<<(DD + 255) / 256, 256, 0, stream>>>(wv, wv_t, D_MODEL, D_MODEL);
  cvt_tr_f16_kernel<<<(DD + 255) / 256, 256, 0, stream>>>(wo, wo_t, D_MODEL, D_MODEL);
  cvt_tr_f16_kernel<<<(DF + 255) / 256, 256, 0, stream>>>(w1, w1_t, D_MODEL, D_FF);
  cvt_tr_f16_kernel<<<(DF + 255) / 256, 256, 0, stream>>>(w2, w2_t, D_FF, D_MODEL);

  // LN1
  ln_f16_kernel<<<MROWS, 256, 0, stream>>>(x, g1, be1, h1, D_MODEL);

  // QKV projections (Q/K: [B,H,S,Dh]; V: [B,H,Dh,S])
  dim3 gproj(D_MODEL / 64, MROWS / 256);
  gemm_wmma_kernel<<<gproj, 256, 0, stream>>>(h1, wq_t, bq, nullptr, qb,
                                              MROWS, D_MODEL, D_MODEL, 1, SEQ);
  gemm_wmma_kernel<<<gproj, 256, 0, stream>>>(h1, wk_t, bk, nullptr, kb,
                                              MROWS, D_MODEL, D_MODEL, 1, SEQ);
  gemm_wmma_kernel<<<gproj, 256, 0, stream>>>(h1, wv_t, bv, nullptr, vt,
                                              MROWS, D_MODEL, D_MODEL, 4, SEQ);

  // attention (flash, WMMA)
  dim3 gattn(BATCH * NHEAD, SEQ / 128);
  flash_attn_kernel<<<gattn, 256, 0, stream>>>(qb, kb, vt, ob, SEQ);

  // output projection + residual (mode 2, f32)
  gemm_wmma_kernel<<<gproj, 256, 0, stream>>>(ob, wo_t, bo, x, x2,
                                              MROWS, D_MODEL, D_MODEL, 2, 0);

  // LN2
  ln_f16_kernel<<<MROWS, 256, 0, stream>>>(x2, g2, be2, h2b, D_MODEL);

  // FFN1 with exact GELU (mode 3, f16 out)
  dim3 gffn1(D_FF / 64, MROWS / 256);
  gemm_wmma_kernel<<<gffn1, 256, 0, stream>>>(h2b, w1_t, b1, nullptr, ffn,
                                              MROWS, D_FF, D_MODEL, 3, 0);

  // FFN2 + residual -> final f32 output (mode 2)
  gemm_wmma_kernel<<<gproj, 256, 0, stream>>>(ffn, w2_t, b2, x2, out,
                                              MROWS, D_MODEL, D_FF, 2, 0);
  (void)in_sizes; (void)n_in; (void)out_size; (void)ws_size;
}
